// MoELayer_1717986918823
// MI455X (gfx1250) — matrix operand order repaired and checked by reading, not compiled, not run on hardware
//
#include <hip/hip_runtime.h>
#include <hip/hip_bf16.h>

typedef __attribute__((ext_vector_type(16))) __bf16 v16bf;
typedef __attribute__((ext_vector_type(8)))  float  v8f;

#define NE 8
#define NH 768
#define NF 3072
#define NT 2048      // B*S tokens
#define XS_STRIDE 776   // 768 + 8 pad (u16 elems), row = 1552B (16B aligned, bank-skewed)
#define HS_STRIDE 264   // 256 + 8 pad (u16 elems), row = 528B  (16B aligned, bank-skewed)

struct Frag32 { uint4 lo; uint4 hi; };   // 32 bytes == v16bf

static __device__ inline unsigned short f2bf(float f) {
    unsigned u = __float_as_uint(f);
    unsigned r = u + 0x7FFFu + ((u >> 16) & 1u);   // round-to-nearest-even
    return (unsigned short)(r >> 16);
}

// A-matrix fragment (16x32 bf16): lane l<16 -> row l, K = k0..k0+15;
// lane l>=16 -> row l-16, K = k0+16..k0+31. Two 16B segments per lane.
static __device__ inline v16bf load_a_frag(const unsigned short* base, int row_stride,
                                           int kbase, int lane) {
    int r = lane & 15, hsel = lane >> 4;
    const unsigned short* p = base + (size_t)r * row_stride + kbase + hsel * 16;
    Frag32 f;
    f.lo = *(const uint4*)p;
    f.hi = *(const uint4*)(p + 8);
    return __builtin_bit_cast(v16bf, f);
}

// B-matrix fragment (32x16 bf16): lane holds column n = l&15, 16 contiguous K values
// starting at (l>>4)*16. p must already point at (row n, k-segment) — 32 contiguous bytes.
static __device__ inline v16bf load_b_frag(const unsigned short* p) {
    Frag32 f;
    f.lo = *(const uint4*)p;
    f.hi = *(const uint4*)(p + 8);
    return __builtin_bit_cast(v16bf, f);
}

static __device__ inline v8f wmma_bf16(v16bf a, v16bf b, v8f c) {
    return __builtin_amdgcn_wmma_f32_16x16x32_bf16(false, a, false, b, (short)0, c,
                                                   false, false);
}

// ---------------------------------------------------------------- fp32 -> bf16
__global__ void cvt_bf16_kernel(const float* __restrict__ src,
                                unsigned short* __restrict__ dst, long n) {
    long i = (long)blockIdx.x * blockDim.x + threadIdx.x;
    long stride = (long)gridDim.x * blockDim.x;
    for (; i < n; i += stride) dst[i] = f2bf(src[i]);
}

// ---------------------------------------------------------------- router
// One wave32 per token: 8 logits, top-2, softmax over the pair, append to
// per-expert compacted lists.
__global__ void router_kernel(const float* __restrict__ x,
                              const float* __restrict__ rw,
                              int* __restrict__ counts,
                              int* __restrict__ lists,
                              float* __restrict__ plist) {
    int wave = threadIdx.x >> 5;
    int lane = threadIdx.x & 31;
    int t = blockIdx.x * 8 + wave;
    if (t >= NT) return;

    const float* xt = x + (size_t)t * NH;
    float lg[NE];
#pragma unroll
    for (int e = 0; e < NE; ++e) {
        const float* we = rw + (size_t)e * NH;
        float s = 0.f;
        for (int i = lane; i < NH; i += 32) s += xt[i] * we[i];
#pragma unroll
        for (int off = 16; off > 0; off >>= 1) s += __shfl_xor(s, off, 32);
        lg[e] = s;   // all lanes hold the full sum
    }
    // top-2 (earliest index wins ties, matching jax.lax.top_k)
    int i0 = 0; float v0 = lg[0];
#pragma unroll
    for (int e = 1; e < NE; ++e) if (lg[e] > v0) { v0 = lg[e]; i0 = e; }
    int i1 = -1; float v1 = -3.4e38f;
#pragma unroll
    for (int e = 0; e < NE; ++e) if (e != i0 && lg[e] > v1) { v1 = lg[e]; i1 = e; }
    float ex = __expf(v1 - v0);          // v1 <= v0, no overflow
    float p0 = 1.f / (1.f + ex);
    float p1 = ex  / (1.f + ex);

    if (lane == 0) {
        int s0 = atomicAdd(&counts[i0], 1);
        lists[(size_t)i0 * NT + s0] = t;  plist[(size_t)i0 * NT + s0] = p0;
        int s1 = atomicAdd(&counts[i1], 1);
        lists[(size_t)i1 * NT + s1] = t;  plist[(size_t)i1 * NT + s1] = p1;
    }
}

// ---------------------------------------------------------------- fused expert FFN
// block = 256 threads (8 waves). blockIdx = (token-tile, expert).
// Wave w: stage-1 computes hmid[16 x 32] for f-slice w*32 of the current 256-wide
// F chunk (full H reduction, no cross-wave reduce), GELU -> LDS; stage-2 consumes
// the shared hmid[16 x 256] against w2, accumulating out[16 x 96] per wave.
__global__ void __launch_bounds__(256)
moe_ffn_kernel(const unsigned short* __restrict__ xbf,
               const unsigned short* __restrict__ w1bf,
               const unsigned short* __restrict__ w2bf,
               const float* __restrict__ b1,
               const float* __restrict__ b2,
               const int* __restrict__ counts,
               const int* __restrict__ lists,
               const float* __restrict__ plist,
               float* __restrict__ out) {
    __shared__ __align__(16) unsigned short xs[16 * XS_STRIDE]; // 24832 B
    __shared__ __align__(16) unsigned short hs[16 * HS_STRIDE]; //  8448 B
    __shared__ int   tk[16];
    __shared__ float tp[16];

    const int e    = blockIdx.y;
    const int tile = blockIdx.x;
    const int cnt  = counts[e];
    if (tile * 16 >= cnt) return;        // uniform exit: EXEC stays all-ones below

    const int tid  = threadIdx.x;
    const int w    = tid >> 5;
    const int lane = tid & 31;
    const int nl   = lane & 15;
    const int hl   = lane >> 4;

    if (tid < 16) {
        int slot = tile * 16 + tid;
        if (slot < cnt) { tk[tid] = lists[(size_t)e * NT + slot];
                          tp[tid] = plist[(size_t)e * NT + slot]; }
        else            { tk[tid] = 0; tp[tid] = 0.f; }   // pad: garbage * 0 at scatter
    }
    __syncthreads();

    // gather x tile: 16 rows x 768 bf16, 16B chunks (96 per row)
    for (int c = tid; c < 16 * 96; c += 256) {
        int r = c / 96, cc = c % 96;
        *(uint4*)(xs + (size_t)r * XS_STRIDE + cc * 8) =
            *(const uint4*)(xbf + (size_t)tk[r] * NH + cc * 8);
    }
    __syncthreads();

    v8f oacc[6];
#pragma unroll
    for (int j = 0; j < 6; ++j) oacc[j] = (v8f){0.f,0.f,0.f,0.f,0.f,0.f,0.f,0.f};

    for (int fbase = 0; fbase < NF; fbase += 256) {
        // ---- stage 1: hmid[16 x 32] for this wave's f-slice
        const int f0 = fbase + w * 32;
        v8f h0 = (v8f){0.f,0.f,0.f,0.f,0.f,0.f,0.f,0.f};
        v8f h1 = h0;
        const unsigned short* w1p =
            w1bf + ((size_t)e * NF + f0 + nl) * NH + hl * 16;
        for (int k = 0; k < NH; k += 32) {
            v16bf a  = load_a_frag(xs, XS_STRIDE, k, lane);
            v16bf bA = load_b_frag(w1p + k);                 // f-tile f0..f0+15
            v16bf bB = load_b_frag(w1p + (size_t)16 * NH + k); // f-tile f0+16..f0+31
            h0 = wmma_bf16(a, bA, h0);
            h1 = wmma_bf16(a, bB, h1);
        }
        // bias + exact GELU -> bf16 -> LDS (C layout: M = v + 8*hl, N = nl)
        float bb0 = b1[(size_t)e * NF + f0 + nl];
        float bb1 = b1[(size_t)e * NF + f0 + 16 + nl];
#pragma unroll
        for (int v = 0; v < 8; ++v) {
            int m = v + 8 * hl;
            float x0 = h0[v] + bb0;
            float x1 = h1[v] + bb1;
            float g0 = 0.5f * x0 * (1.f + erff(x0 * 0.70710678118654752f));
            float g1 = 0.5f * x1 * (1.f + erff(x1 * 0.70710678118654752f));
            hs[(size_t)m * HS_STRIDE + w * 32 + nl]      = f2bf(g0);
            hs[(size_t)m * HS_STRIDE + w * 32 + 16 + nl] = f2bf(g1);
        }
        __syncthreads();

        // ---- stage 2: out[16 x 96] += hmid[16 x 256] @ w2[e]^T slice
        for (int k2 = 0; k2 < 256; k2 += 32) {
            v16bf a = load_a_frag(hs, HS_STRIDE, k2, lane);
#pragma unroll
            for (int j = 0; j < 6; ++j) {
                const unsigned short* w2p =
                    w2bf + ((size_t)e * NH + w * 96 + j * 16 + nl) * NF
                         + fbase + k2 + hl * 16;
                oacc[j] = wmma_bf16(a, load_b_frag(w2p), oacc[j]);
            }
        }
        __syncthreads();   // hmid consumed before next chunk overwrites it
    }

    // ---- epilogue: +b2, scale by routing prob, scatter-accumulate
#pragma unroll
    for (int j = 0; j < 6; ++j) {
        int hcol = w * 96 + j * 16 + nl;
        float b2v = b2[(size_t)e * NH + hcol];
#pragma unroll
        for (int v = 0; v < 8; ++v) {
            int m = v + 8 * hl;
            float val = (oacc[j][v] + b2v) * tp[m];
            atomicAdd(out + (size_t)tk[m] * NH + hcol, val);
        }
    }
}

// ---------------------------------------------------------------- launch
extern "C" void kernel_launch(void* const* d_in, const int* in_sizes, int n_in,
                              void* d_out, int out_size, void* d_ws, size_t ws_size,
                              hipStream_t stream) {
    const float* x  = (const float*)d_in[0];   // [B,S,H]
    const float* rw = (const float*)d_in[1];   // [E,H]
    const float* w1 = (const float*)d_in[2];   // [E,F,H]
    const float* b1 = (const float*)d_in[3];   // [E,F]
    const float* w2 = (const float*)d_in[4];   // [E,H,F]
    const float* b2 = (const float*)d_in[5];   // [E,H]
    float* out = (float*)d_out;

    char* ws = (char*)d_ws;
    size_t off = 0;
    auto carve = [&](size_t bytes) -> void* {
        void* p = ws + off;
        off = (off + bytes + 255) & ~(size_t)255;
        return p;
    };
    unsigned short* xbf  = (unsigned short*)carve((size_t)NT * NH * 2);       // ~3.1 MB
    unsigned short* w1bf = (unsigned short*)carve((size_t)NE * NF * NH * 2);  // ~37.7 MB
    unsigned short* w2bf = (unsigned short*)carve((size_t)NE * NH * NF * 2);  // ~37.7 MB
    int*   counts = (int*)  carve((size_t)NE * sizeof(int));
    int*   lists  = (int*)  carve((size_t)NE * NT * sizeof(int));
    float* plist  = (float*)carve((size_t)NE * NT * sizeof(float));

    hipMemsetAsync(counts, 0, (size_t)NE * sizeof(int), stream);
    hipMemsetAsync(out, 0, (size_t)out_size * sizeof(float), stream);

    cvt_bf16_kernel<<<1024, 256, 0, stream>>>(x,  xbf,  (long)NT * NH);
    cvt_bf16_kernel<<<4096, 256, 0, stream>>>(w1, w1bf, (long)NE * NF * NH);
    cvt_bf16_kernel<<<4096, 256, 0, stream>>>(w2, w2bf, (long)NE * NH * NF);

    router_kernel<<<NT / 8, 256, 0, stream>>>(x, rw, counts, lists, plist);

    moe_ffn_kernel<<<dim3(128, NE), 256, 0, stream>>>(
        xbf, w1bf, w2bf, b1, b2, counts, lists, plist, out);
}
